// HierarchicalAttentionFusion_73624329388234
// MI455X (gfx1250) — compile-verified, matrix operand order checked by reading, not run on hardware
//
#include <hip/hip_runtime.h>
#include <math.h>

#define BN 4096
#define DF 256

#ifndef __has_builtin
#define __has_builtin(x) 0
#endif

#if __has_builtin(__builtin_amdgcn_global_load_async_to_lds_b128) && \
    __has_builtin(__builtin_amdgcn_s_wait_asynccnt)
#define USE_ASYNC_LDS 1
typedef int v4i_ag __attribute__((vector_size(16)));
typedef __attribute__((address_space(1))) v4i_ag* gv4_t;   // global int4*
typedef __attribute__((address_space(3))) v4i_ag* lv4_t;   // LDS int4*
#else
#define USE_ASYNC_LDS 0
#endif

typedef __attribute__((ext_vector_type(16))) __bf16 v16bf;
typedef __attribute__((ext_vector_type(8)))  float  v8f;

// native f32 -> bf16 converts (let the compiler emit v_cvt_*bf16_f32)
__device__ __forceinline__ unsigned short f2bf(float f) {
  union { __bf16 h; unsigned short u; } p;
  p.h = (__bf16)f;
  return p.u;
}
__device__ __forceinline__ unsigned int pk2(float a, float b) {
  union { __bf16 h[2]; unsigned int u; } p;
  p.h[0] = (__bf16)a; p.h[1] = (__bf16)b;
  return p.u;
}
__device__ __forceinline__ float bf2f(unsigned short h) {
  return __uint_as_float(((unsigned int)h) << 16);
}
__device__ __forceinline__ float gelu_f(float x) {
  return 0.5f * x * (1.0f + erff(x * 0.7071067811865475f));
}

union Frag { uint4 q[2]; v16bf v; };

// C = act(A @ B + bias) [* rscale[m]]   (row-major A [M,K], B [K,N] or Bt [N,K])
// KM epilogue: write bf16 exp(-sqrt(max(ssq[m]-2*acc+tsq[n],0))/temp) instead.
template<int A_BF16, int TRANS_B, int ACT, int KM>
__global__ __launch_bounds__(256) void gemm64(
    const void* __restrict__ Ap, const float* __restrict__ Bp,
    const float* __restrict__ bias, const float* __restrict__ rscale,
    const float* __restrict__ ssq, const float* __restrict__ tsq,
    const float* __restrict__ temperature,
    void* __restrict__ Cp, int M, int N, int K)
{
  (void)M;
  __shared__ __align__(16) unsigned short As[64 * 40];
  __shared__ __align__(16) unsigned short Bs[64 * 40];

  const int tid  = threadIdx.x;
  const int lane = tid & 31;
  const int wave = tid >> 5;
  const int lr   = lane & 15;
  const int half = lane >> 4;
  const int wm   = wave & 3;   // 4 M-subtiles of 16
  const int wn   = wave >> 2;  // 2 N-groups of 32
  const int m0   = blockIdx.y * 64;
  const int n0   = blockIdx.x * 64;

  v8f acc0 = {0.f, 0.f, 0.f, 0.f, 0.f, 0.f, 0.f, 0.f};
  v8f acc1 = acc0;

  const int aidx = tid * 8;
  const int ar_ = aidx >> 5, ac_ = aidx & 31;   // A/Bt staging coords
  const int kk_ = aidx >> 6, nb_ = aidx & 63;   // B (non-trans) staging coords

  for (int k0 = 0; k0 < K; k0 += 32) {
    // ---- stage A tile (64 x 32) as bf16, row-major stride 40 ----
    if (A_BF16) {
      const unsigned short* src = (const unsigned short*)Ap + (size_t)(m0 + ar_) * K + k0 + ac_;
#if USE_ASYNC_LDS
      // CDNA5 async copy: bytes go straight to per-lane LDS dest, tracked by ASYNCcnt
      __builtin_amdgcn_global_load_async_to_lds_b128(
          (gv4_t)src, (lv4_t)&As[ar_ * 40 + ac_], 0, 0);
#else
      *(uint4*)&As[ar_ * 40 + ac_] = *(const uint4*)src;
#endif
      if (k0 + 32 < K) __builtin_prefetch(src + 32, 0, 1);
    } else {
      const float* src = (const float*)Ap + (size_t)(m0 + ar_) * K + k0 + ac_;
      float4 f0 = *(const float4*)src;
      float4 f1 = *(const float4*)(src + 4);
      uint4 q; q.x = pk2(f0.x, f0.y); q.y = pk2(f0.z, f0.w);
      q.z = pk2(f1.x, f1.y); q.w = pk2(f1.z, f1.w);
      *(uint4*)&As[ar_ * 40 + ac_] = q;
      if (k0 + 32 < K) __builtin_prefetch(src + 32, 0, 1);
    }
    // ---- stage B tile into Bs[n][k] (K-contiguous per output column) ----
    if (TRANS_B) {
      const float* src = (const float*)Bp + (size_t)(n0 + ar_) * K + k0 + ac_;
      float4 f0 = *(const float4*)src;
      float4 f1 = *(const float4*)(src + 4);
      uint4 q; q.x = pk2(f0.x, f0.y); q.y = pk2(f0.z, f0.w);
      q.z = pk2(f1.x, f1.y); q.w = pk2(f1.z, f1.w);
      *(uint4*)&Bs[ar_ * 40 + ac_] = q;
    } else {
      const float* src = (const float*)Bp + (size_t)(k0 + kk_) * N + n0 + nb_;
      float4 f0 = *(const float4*)src;
      float4 f1 = *(const float4*)(src + 4);
      float vals[8] = {f0.x, f0.y, f0.z, f0.w, f1.x, f1.y, f1.z, f1.w};
#pragma unroll
      for (int i = 0; i < 8; ++i) Bs[(nb_ + i) * 40 + kk_] = f2bf(vals[i]);
    }
#if USE_ASYNC_LDS
    if (A_BF16) __builtin_amdgcn_s_wait_asynccnt(0);
#endif
    __syncthreads();

    // ---- fragments per ISA 16-bit layout: lane<16 K{0..7,16..23}, lane>=16 K{8..15,24..31} ----
    Frag fA, fB0, fB1;
    const unsigned short* arow = &As[(wm * 16 + lr) * 40];
    fA.q[0] = *(const uint4*)(arow + half * 8);
    fA.q[1] = *(const uint4*)(arow + 16 + half * 8);
    const unsigned short* br0 = &Bs[(wn * 32 + lr) * 40];
    fB0.q[0] = *(const uint4*)(br0 + half * 8);
    fB0.q[1] = *(const uint4*)(br0 + 16 + half * 8);
    const unsigned short* br1 = &Bs[(wn * 32 + 16 + lr) * 40];
    fB1.q[0] = *(const uint4*)(br1 + half * 8);
    fB1.q[1] = *(const uint4*)(br1 + 16 + half * 8);

    acc0 = __builtin_amdgcn_wmma_f32_16x16x32_bf16(false, fA.v, false, fB0.v, (short)0, acc0, false, false);
    acc1 = __builtin_amdgcn_wmma_f32_16x16x32_bf16(false, fA.v, false, fB1.v, (short)0, acc1, false, false);
    __syncthreads();
  }

  const float tinv = KM ? (1.0f / temperature[0]) : 0.0f;
#pragma unroll
  for (int j = 0; j < 2; ++j) {
    v8f a = j ? acc1 : acc0;
    const int n = n0 + wn * 32 + j * 16 + lr;
#pragma unroll
    for (int r = 0; r < 8; ++r) {
      const int m = m0 + wm * 16 + half * 8 + r;
      float v = a[r];
      if (bias) v += bias[n];
      if (ACT == 1) v = gelu_f(v);
      if (KM) {
        float d2 = ssq[m] - 2.0f * v + tsq[n];
        float cst = sqrtf(fmaxf(d2, 0.0f));
        float km = expf(-cst * tinv);
        ((unsigned short*)Cp)[(size_t)m * N + n] = f2bf(km);
      } else {
        if (rscale) v *= rscale[m];
        ((float*)Cp)[(size_t)m * N + n] = v;
      }
    }
  }
}

// ---------------- small glue kernels ----------------

__device__ __forceinline__ float bredN(float* red, int t, int n, float v) {
  red[t] = v; __syncthreads();
  for (int off = n >> 1; off > 0; off >>= 1) {
    if (t < off) red[t] += red[t + off];
    __syncthreads();
  }
  float s = red[0]; __syncthreads();
  return s;
}

__global__ void gate_top2_k(const float* __restrict__ H, const float* __restrict__ W2,
                            const float* __restrict__ b2, float* __restrict__ dw, int h) {
  __shared__ float red[4][128];
  const int b = blockIdx.x, t = threadIdx.x;
  float s0 = 0, s1 = 0, s2 = 0, s3 = 0;
  for (int i = t; i < h; i += 128) {
    float x = H[(size_t)b * h + i];
    const float* w = W2 + (size_t)i * 4;
    s0 += x * w[0]; s1 += x * w[1]; s2 += x * w[2]; s3 += x * w[3];
  }
  red[0][t] = s0; red[1][t] = s1; red[2][t] = s2; red[3][t] = s3;
  __syncthreads();
  for (int off = 64; off > 0; off >>= 1) {
    if (t < off) {
      red[0][t] += red[0][t + off]; red[1][t] += red[1][t + off];
      red[2][t] += red[2][t + off]; red[3][t] += red[3][t + off];
    }
    __syncthreads();
  }
  if (t == 0) {
    float sc[4];
    for (int e = 0; e < 4; ++e) sc[e] = red[e][0] + b2[e];
    int i0 = 0;
    for (int e = 1; e < 4; ++e) if (sc[e] > sc[i0]) i0 = e;
    int i1 = -1;
    for (int e = 0; e < 4; ++e) { if (e == i0) continue; if (i1 < 0 || sc[e] > sc[i1]) i1 = e; }
    float e1 = expf(sc[i1] - sc[i0]);
    float z = 1.0f + e1;
    float* o = dw + (size_t)b * 4;
    o[0] = 0.f; o[1] = 0.f; o[2] = 0.f; o[3] = 0.f;
    o[i0] = 1.0f / z; o[i1] = e1 / z;
  }
}

__global__ void moe_combine_k(const float* __restrict__ EO, const float* __restrict__ dw,
                              const float* __restrict__ lng, const float* __restrict__ lnb,
                              float* __restrict__ out, int nB) {
  __shared__ float red[256];
  const int b = blockIdx.x, t = threadIdx.x;
  float acc = 0.f;
  for (int e = 0; e < 4; ++e) {
    float x = EO[((size_t)e * nB + b) * DF + t];
    float mu = bredN(red, t, 256, x) * (1.0f / DF);
    float d = x - mu;
    float var = bredN(red, t, 256, d * d) * (1.0f / DF);
    float y = d * rsqrtf(var + 1e-5f) * lng[e * DF + t] + lnb[e * DF + t];
    acc += dw[(size_t)b * 4 + e] * gelu_f(y);
  }
  out[(size_t)b * DF + t] = acc;
}

__global__ void ln_act_row_k(const float* __restrict__ X, const float* __restrict__ g,
                             const float* __restrict__ bb, float* __restrict__ Y,
                             int n, int act) {
  extern __shared__ float red[];
  const int b = blockIdx.x, t = threadIdx.x;
  float x = X[(size_t)b * n + t];
  float mu = bredN(red, t, n, x) / (float)n;
  float d = x - mu;
  float var = bredN(red, t, n, d * d) / (float)n;
  float y = d * rsqrtf(var + 1e-5f) * g[t] + bb[t];
  y = (act == 0) ? fmaxf(y, 0.f) : gelu_f(y);
  Y[(size_t)b * n + t] = y;
}

__global__ void sigmoid_mul_k(const float* __restrict__ x, const float* __restrict__ y,
                              float* __restrict__ o, int n) {
  int i = blockIdx.x * blockDim.x + threadIdx.x;
  if (i < n) o[i] = x[i] * (1.0f / (1.0f + expf(-y[i])));
}

__global__ void rownorm2_k(const float* __restrict__ X, float* __restrict__ o) {
  __shared__ float red[256];
  int b = blockIdx.x, t = threadIdx.x;
  float x = X[(size_t)b * DF + t];
  float s = bredN(red, t, 256, x * x);
  if (t == 0) o[b] = s;
}

__global__ void rowsum_recip_k(const unsigned short* __restrict__ Km, float* __restrict__ rrec, int n) {
  __shared__ float red[256];
  int b = blockIdx.x, t = threadIdx.x;
  float s = 0.f;
  for (int j = t; j < n; j += 256) s += bf2f(Km[(size_t)b * n + j]);
  s = bredN(red, t, 256, s);
  if (t == 0) rrec[b] = 1.0f / s;
}

__global__ void concat3_k(const float* __restrict__ fv, const float* __restrict__ fa,
                          const float* __restrict__ ft, float* __restrict__ c, int total) {
  int i = blockIdx.x * blockDim.x + threadIdx.x;
  if (i >= total) return;
  int b = i / 768, j = i - b * 768;
  float v;
  if (j < 256) v = fv[b * 256 + j];
  else if (j < 512) v = fa[b * 256 + j - 256];
  else v = ft[b * 256 + j - 512];
  c[i] = v;
}

__global__ void w3softmax_k(const float* __restrict__ H, const float* __restrict__ W3,
                            const float* __restrict__ b3, float* __restrict__ wts) {
  __shared__ float red[3][128];
  int b = blockIdx.x, t = threadIdx.x;
  float x = H[(size_t)b * 128 + t];
  const float* w = W3 + (size_t)t * 3;
  red[0][t] = x * w[0]; red[1][t] = x * w[1]; red[2][t] = x * w[2];
  __syncthreads();
  for (int off = 64; off > 0; off >>= 1) {
    if (t < off) {
      red[0][t] += red[0][t + off]; red[1][t] += red[1][t + off]; red[2][t] += red[2][t + off];
    }
    __syncthreads();
  }
  if (t == 0) {
    float a0 = red[0][0] + b3[0], a1 = red[1][0] + b3[1], a2 = red[2][0] + b3[2];
    float m = fmaxf(a0, fmaxf(a1, a2));
    float e0 = expf(a0 - m), e1 = expf(a1 - m), e2 = expf(a2 - m);
    float z = e0 + e1 + e2;
    wts[(size_t)b * 3 + 0] = e0 / z; wts[(size_t)b * 3 + 1] = e1 / z; wts[(size_t)b * 3 + 2] = e2 / z;
  }
}

__global__ void fusemix_k(const float* __restrict__ fv, const float* __restrict__ fa,
                          const float* __restrict__ ft, const float* __restrict__ wts,
                          float* __restrict__ o, int total) {
  int i = blockIdx.x * blockDim.x + threadIdx.x;
  if (i >= total) return;
  int b = i >> 8;
  o[i] = fv[i] * wts[b * 3] + fa[i] * wts[b * 3 + 1] + ft[i] * wts[b * 3 + 2];
}

// ---------------- host orchestration ----------------

extern "C" void kernel_launch(void* const* d_in, const int* in_sizes, int n_in,
                              void* d_out, int out_size, void* d_ws, size_t ws_size,
                              hipStream_t stream) {
  (void)in_sizes; (void)n_in; (void)out_size; (void)ws_size;
  const int Bb = BN;

  size_t off = 0;
  auto alloc = [&](size_t nf) -> float* { float* p = (float*)d_ws + off; off += nf; return p; };
  float* fv   = alloc((size_t)Bb * DF);
  float* fa   = alloc((size_t)Bb * DF);
  float* ft   = alloc((size_t)Bb * DF);
  float* H1   = alloc((size_t)Bb * 512);
  float* EO   = alloc((size_t)4 * Bb * DF);
  float* dw   = alloc((size_t)Bb * 4);
  float* moe  = alloc((size_t)Bb * DF);
  float* t1   = alloc((size_t)Bb * DF);
  float* t2   = alloc((size_t)Bb * DF);
  float* sp   = alloc((size_t)Bb * DF);
  float* tp   = alloc((size_t)Bb * DF);
  float* ss   = alloc((size_t)Bb);
  float* tt   = alloc((size_t)Bb);
  float* rrec = alloc((size_t)Bb);
  float* cbuf = alloc((size_t)Bb * 768);
  float* wts  = alloc((size_t)Bb * 4);
  unsigned short* Km = (unsigned short*)alloc((size_t)Bb * (size_t)Bb / 2);

  float* outFused = (float*)d_out;
  float* outVA = outFused + (size_t)Bb * DF;
  float* outVT = outVA + (size_t)Bb * DF;
  float* outAT = outVT + (size_t)Bb * DF;

  auto gemm = [&](const float* A, const float* Bm, const float* bias, float* C,
                  int N, int K, int act) {
    dim3 g(N / 64, Bb / 64);
    if (act == 1)
      gemm64<0, 0, 1, 0><<<g, 256, 0, stream>>>(A, Bm, bias, nullptr, nullptr, nullptr, nullptr, C, Bb, N, K);
    else
      gemm64<0, 0, 0, 0><<<g, 256, 0, stream>>>(A, Bm, bias, nullptr, nullptr, nullptr, nullptr, C, Bb, N, K);
  };

  auto modality = [&](const float* x, int e_in, int pbase, float* fout) {
    const float* gw1 = (const float*)d_in[pbase + 0];
    const float* gb1 = (const float*)d_in[pbase + 1];
    const float* gw2 = (const float*)d_in[pbase + 2];
    const float* gb2 = (const float*)d_in[pbase + 3];
    const float* ew  = (const float*)d_in[pbase + 4];
    const float* eb  = (const float*)d_in[pbase + 5];
    const float* lng = (const float*)d_in[pbase + 6];
    const float* lnb = (const float*)d_in[pbase + 7];
    const float* sw1 = (const float*)d_in[pbase + 8];
    const float* sb1 = (const float*)d_in[pbase + 9];
    const float* sg  = (const float*)d_in[pbase + 10];
    const float* sbb = (const float*)d_in[pbase + 11];
    const float* sw2 = (const float*)d_in[pbase + 12];
    const float* sb2 = (const float*)d_in[pbase + 13];
    const int h = e_in / 2;
    gemm(x, gw1, gb1, H1, h, e_in, 1);                       // gate hidden + GELU
    gate_top2_k<<<Bb, 128, 0, stream>>>(H1, gw2, gb2, dw, h);
    for (int e = 0; e < 4; ++e)                              // dense experts
      gemm(x, ew + (size_t)e * e_in * DF, eb + (size_t)e * DF, EO + (size_t)e * Bb * DF, DF, e_in, 0);
    moe_combine_k<<<Bb, 256, 0, stream>>>(EO, dw, lng, lnb, moe, Bb);
    gemm(moe, sw1, sb1, t1, 128, DF, 0);                     // feature gating
    ln_act_row_k<<<Bb, 128, 128 * sizeof(float), stream>>>(t1, sg, sbb, t1, 128, 0);
    gemm(t1, sw2, sb2, t2, DF, 128, 0);
    int tot = Bb * DF;
    sigmoid_mul_k<<<(tot + 255) / 256, 256, 0, stream>>>(moe, t2, fout, tot);
  };

  modality((const float*)d_in[0], 1024, 3, fv);
  modality((const float*)d_in[1], 768, 17, fa);
  modality((const float*)d_in[2], 768, 31, ft);

  auto align_pair = [&](const float* s, const float* t, int pbase, float* out) {
    const float* swp = (const float*)d_in[pbase + 0];
    const float* sbp = (const float*)d_in[pbase + 1];
    const float* twp = (const float*)d_in[pbase + 2];
    const float* tbp = (const float*)d_in[pbase + 3];
    const float* temp = (const float*)d_in[pbase + 4];
    gemm(s, swp, sbp, sp, DF, DF, 0);
    gemm(t, twp, tbp, tp, DF, DF, 0);
    rownorm2_k<<<Bb, 256, 0, stream>>>(sp, ss);
    rownorm2_k<<<Bb, 256, 0, stream>>>(tp, tt);
    {
      dim3 g(Bb / 64, Bb / 64);   // Km = exp(-cdist/temp) fused into sp @ tp^T epilogue (bf16 out)
      gemm64<0, 1, 0, 1><<<g, 256, 0, stream>>>(sp, tp, nullptr, nullptr, ss, tt, temp, Km, Bb, Bb, DF);
    }
    rowsum_recip_k<<<Bb, 256, 0, stream>>>(Km, rrec, Bb);
    {
      dim3 g(DF / 64, Bb / 64);   // (Km @ t) * (1/rowsum), A is bf16 (async-LDS staging path)
      gemm64<1, 0, 0, 0><<<g, 256, 0, stream>>>(Km, t, nullptr, rrec, nullptr, nullptr, nullptr, out, Bb, DF, Bb);
    }
  };

  align_pair(fv, fa, 45, outVA);
  align_pair(fv, ft, 50, outVT);
  align_pair(fa, ft, 55, outAT);

  {
    const float* w1 = (const float*)d_in[60]; const float* b1 = (const float*)d_in[61];
    const float* g  = (const float*)d_in[62]; const float* bb = (const float*)d_in[63];
    const float* w2 = (const float*)d_in[64]; const float* b2 = (const float*)d_in[65];
    const float* w3 = (const float*)d_in[66]; const float* b3 = (const float*)d_in[67];
    int tot3 = Bb * 768;
    concat3_k<<<(tot3 + 255) / 256, 256, 0, stream>>>(fv, fa, ft, cbuf, tot3);
    gemm(cbuf, w1, b1, t1, DF, 768, 0);
    ln_act_row_k<<<Bb, 256, 256 * sizeof(float), stream>>>(t1, g, bb, t1, DF, 1);
    gemm(t1, w2, b2, t2, 128, DF, 1);
    w3softmax_k<<<Bb, 128, 0, stream>>>(t2, w3, b3, wts);
    int tot = Bb * DF;
    fusemix_k<<<(tot + 255) / 256, 256, 0, stream>>>(fv, fa, ft, wts, outFused, tot);
  }
}
